// GaussianKernel_48490180771969
// MI455X (gfx1250) — matrix operand order repaired and checked by reading, not compiled, not run on hardware
//
#include <hip/hip_runtime.h>
#include <hip/hip_bf16.h>

// ---------------------------------------------------------------------------
// RBF kernel layer:  out = exp(-gamma * ||x - w||_2) + b
// via GEMM expansion ||x-w||^2 = x^2 - 2 x.w + w^2, fused epilogue.
//   x: (64,56,56,64) f32 -> rows N=200704, K=64
//   w: (64,128) f32      -> C_out = 128
// bf16 WMMA (v_wmma_f32_16x16x32_bf16) for the cross term, f32 norms/epilogue.
// Each wave: 2 M-tiles (32 rows) x 8 N-tiles x K=64  -> 32 WMMAs / wave.
// Column-tile loop at unroll 2 to cap accumulator liveness (occupancy).
// Epilogue uses raw v_sqrt_f32 / v_exp_f32 (args never in fixup ranges).
// ---------------------------------------------------------------------------

typedef __attribute__((ext_vector_type(16))) __bf16 v16bf;
typedef __attribute__((ext_vector_type(8)))  float  v8f;

#define CIN        64
#define COUT       128
#define WT_STRIDE  72       // padded bf16 column stride (144 B, 16B-aligned)
#define MT         2        // M-tiles (of 16 rows) per wave
#define WAVES      8
#define ROWS_PER_BLOCK (WAVES * 16 * MT)   // 256

union ABFrag {
    v16bf v;
    uint4 q[2];
};

__device__ __forceinline__ unsigned short f32_to_bf16_rne(float f) {
    unsigned int u = __float_as_uint(f);
    unsigned int r = u + 0x7FFFu + ((u >> 16) & 1u);
    return (unsigned short)(r >> 16);
}

// ---------------------------------------------------------------------------
// Prep: wT[o*WT_STRIDE + c] = bf16(w[c][o]);  w_sq[o] = sum_c w[c][o]^2
// ---------------------------------------------------------------------------
__global__ void rbf_prep_kernel(const float* __restrict__ w,
                                unsigned short* __restrict__ wT,
                                float* __restrict__ w_sq) {
    int o = threadIdx.x;          // 0..127
    float acc = 0.f;
#pragma unroll
    for (int c = 0; c < CIN; ++c) {
        float v = w[c * COUT + o];
        acc += v * v;
        wT[o * WT_STRIDE + c] = f32_to_bf16_rne(v);
    }
    w_sq[o] = acc;
}

// ---------------------------------------------------------------------------
// Main fused kernel.
// ---------------------------------------------------------------------------
__global__ void __launch_bounds__(256)
rbf_main_kernel(const float* __restrict__ x,
                const float* __restrict__ bvec,
                const float* __restrict__ gamma,
                const unsigned short* __restrict__ wT_g,
                const float* __restrict__ wsq_g,
                float* __restrict__ out) {
    // ---- LDS staging of w (bf16, padded), w_sq, b -------------------------
    __shared__ uint4 sWTq[(COUT * WT_STRIDE) / 8];   // 18432 B
    __shared__ float sWSQ[COUT];
    __shared__ float sB[COUT];

    const int tid = threadIdx.x;
    {
        const uint4* src = (const uint4*)wT_g;
        for (int i = tid; i < (COUT * WT_STRIDE) / 8; i += 256)
            sWTq[i] = src[i];
        if (tid < COUT) {
            sWSQ[tid] = wsq_g[tid];
            sB[tid]   = bvec[tid];
        }
    }
    __syncthreads();
    const unsigned short* sWT = (const unsigned short*)sWTq;

    const int wave    = tid >> 5;
    const int lane    = tid & 31;
    const int halfsel = lane >> 4;     // 0: lanes 0-15, 1: lanes 16-31
    const int mrow    = lane & 15;

    const long long wave_row0 = (long long)blockIdx.x * ROWS_PER_BLOCK + wave * (16 * MT);

    // ---- A fragments (ISA 16-bit A 16x32 layout) + f32 row norms ----------
    // lane<16 : K in {kb*32 + 0..7, kb*32 + 16..23}
    // lane>=16: K in {kb*32 + 8..15, kb*32 + 24..31}
    ABFrag a[MT][2];
    float  xs[MT][8];
#pragma unroll
    for (int r = 0; r < MT; ++r) {
        const float* rp = x + (wave_row0 + r * 16 + mrow) * CIN;
        float partial = 0.f;
#pragma unroll
        for (int kb = 0; kb < 2; ++kb) {
            const int base = kb * 32 + halfsel * 8;
            float4 f0 = *(const float4*)(rp + base + 0);
            float4 f1 = *(const float4*)(rp + base + 4);
            float4 f2 = *(const float4*)(rp + base + 16);
            float4 f3 = *(const float4*)(rp + base + 20);
            partial += f0.x*f0.x + f0.y*f0.y + f0.z*f0.z + f0.w*f0.w;
            partial += f1.x*f1.x + f1.y*f1.y + f1.z*f1.z + f1.w*f1.w;
            partial += f2.x*f2.x + f2.y*f2.y + f2.z*f2.z + f2.w*f2.w;
            partial += f3.x*f3.x + f3.y*f3.y + f3.z*f3.z + f3.w*f3.w;
            v16bf av;
            av[0]  = (__bf16)f0.x; av[1]  = (__bf16)f0.y;
            av[2]  = (__bf16)f0.z; av[3]  = (__bf16)f0.w;
            av[4]  = (__bf16)f1.x; av[5]  = (__bf16)f1.y;
            av[6]  = (__bf16)f1.z; av[7]  = (__bf16)f1.w;
            av[8]  = (__bf16)f2.x; av[9]  = (__bf16)f2.y;
            av[10] = (__bf16)f2.z; av[11] = (__bf16)f2.w;
            av[12] = (__bf16)f3.x; av[13] = (__bf16)f3.y;
            av[14] = (__bf16)f3.z; av[15] = (__bf16)f3.w;
            a[r][kb].v = av;
        }
        // lanes L and L^16 both hold the full norm of row (L&15)
        float full = partial + __shfl_xor(partial, 16, 32);
        // broadcast: lane needs rows m = i + 8*halfsel (hoisted out of tile loop)
#pragma unroll
        for (int i = 0; i < 8; ++i)
            xs[r][i] = __shfl(full, halfsel * 8 + i, 32);
    }

    const float ng2 = -gamma[0] * 1.44269504088896340736f;  // -gamma * log2(e)

    // ---- 8 column tiles of 16: two K=32 WMMAs per M-tile ------------------
    // unroll 2: enough independent WMMA/ds work for the scheduler, but caps
    // accumulator liveness so occupancy stays high (memory-bound kernel).
#pragma unroll 2
    for (int t = 0; t < 8; ++t) {
        const int col = t * 16 + mrow;
        const unsigned short* bp = sWT + col * WT_STRIDE;

        // B fragment (16-bit B, K x N): lanes 0-15 hold K={0..7,16..23} (per kb),
        // lanes 16-31 hold K={8..15,24..31}; two 8-bf16 chunks per lane from LDS.
        ABFrag b0, b1;
        b0.q[0] = *(const uint4*)(bp + halfsel * 8 + 0);
        b0.q[1] = *(const uint4*)(bp + halfsel * 8 + 16);
        b1.q[0] = *(const uint4*)(bp + 32 + halfsel * 8 + 0);
        b1.q[1] = *(const uint4*)(bp + 32 + halfsel * 8 + 16);

        const float wsq = sWSQ[col];
        const float bb  = sB[col];

#pragma unroll
        for (int r = 0; r < MT; ++r) {
            v8f c = {};
            c = __builtin_amdgcn_wmma_f32_16x16x32_bf16(
                    false, a[r][0].v, false, b0.v, (short)0, c, false, false);
            c = __builtin_amdgcn_wmma_f32_16x16x32_bf16(
                    false, a[r][1].v, false, b1.v, (short)0, c, false, false);

            // C/D layout: c[i] is row (i + 8*halfsel), column = col
            float* op = out + (wave_row0 + r * 16 + halfsel * 8) * COUT + col;
#pragma unroll
            for (int i = 0; i < 8; ++i) {
                float s  = xs[r][i] + wsq;
                float sq = fmaxf(__builtin_fmaf(-2.0f, c[i], s), 0.0f);
                // raw v_sqrt_f32 / v_exp_f32: args never hit the denormal
                // fixup ranges the libm wrappers guard against.
                float dn = __builtin_amdgcn_sqrtf(sq);
                op[i * COUT] = __builtin_amdgcn_exp2f(ng2 * dn) + bb;
            }
        }
    }
}

// ---------------------------------------------------------------------------
// Host-side launcher (all on `stream`; graph-capture safe).
// ---------------------------------------------------------------------------
extern "C" void kernel_launch(void* const* d_in, const int* in_sizes, int n_in,
                              void* d_out, int out_size, void* d_ws, size_t ws_size,
                              hipStream_t stream) {
    const float* x     = (const float*)d_in[0];
    const float* w     = (const float*)d_in[1];
    const float* b     = (const float*)d_in[2];
    const float* gamma = (const float*)d_in[3];
    float* out = (float*)d_out;

    // workspace: [0,18432) wT bf16 (padded col-major); [18432,18944) w_sq f32
    unsigned short* wT  = (unsigned short*)d_ws;
    float*          wsq = (float*)((char*)d_ws + (size_t)COUT * WT_STRIDE * sizeof(unsigned short));

    const long long n_rows = (long long)in_sizes[0] / CIN;       // 200704
    const int n_blocks     = (int)(n_rows / ROWS_PER_BLOCK);     // 784 (exact)

    rbf_prep_kernel<<<1, COUT, 0, stream>>>(w, wT, wsq);
    rbf_main_kernel<<<n_blocks, 256, 0, stream>>>(x, b, gamma, wT, wsq, out);
    (void)n_in; (void)out_size; (void)ws_size;
}
